// Encoder_31791347925541
// MI455X (gfx1250) — compile-verified, matrix operand order checked by reading, not run on hardware
//
#include <hip/hip_runtime.h>
#include <hip/hip_bf16.h>

typedef __attribute__((ext_vector_type(16))) _Float16 v16h;
typedef __attribute__((ext_vector_type(8)))  float    v8f;

// ---------------- problem constants ----------------
static constexpr int BATCH = 8;
static constexpr int NTOK  = 1024;   // tokens per image (32x32 patches)
static constexpr int DMODEL = 768;
static constexpr int HEADS  = 12;
static constexpr int DH     = 64;
static constexpr int MLPDIM = 3072;
static constexpr int DEPTH  = 8;
static constexpr int KPATCH = 192;   // C*P*P = 3*8*8
static constexpr int MROWS  = BATCH * NTOK;  // 8192

// attention LDS: S block 64 x 1024 fp32, padded stride 1028 (1028 % 64 == 4 -> conflict-free
// column walks), plus two 256-float reduction arrays.
static constexpr int SLD = 1028;
static constexpr int ATTN_LDS_BYTES = (64 * SLD + 512) * (int)sizeof(float);

// ---------------- CDNA5 async copy helpers ----------------
// global_load_async_to_lds_b128 (GVS form): mem = SADDR(u64) + VADDR(i32) ; LDS dst = VDST vgpr.
// Tracked by ASYNCcnt (CDNA5 ISA); completion via s_wait_asynccnt.
typedef __attribute__((address_space(3))) const void* lds_cvptr_t;

__device__ __forceinline__ uint32_t lds_off(const void* p) {
  // generic -> LDS address space -> 32-bit LDS byte offset
  return (uint32_t)(size_t)(lds_cvptr_t)p;
}

__device__ __forceinline__ void async_ld_b128(uint32_t lds_addr, const void* sbase,
                                              uint32_t voff) {
  asm volatile("global_load_async_to_lds_b128 %0, %1, %2"
               :
               : "v"(lds_addr), "v"(voff),
                 "s"((unsigned long long)(size_t)sbase)
               : "memory");
}

__device__ __forceinline__ void wait_async0() {
  asm volatile("s_wait_asynccnt 0" ::: "memory");
}

// ---------------- WMMA helpers (layouts per CDNA5 ISA 7.12.2) ----------------
__device__ __forceinline__ v8f wmma16x16x32(v16h a, v16h b, v8f c) {
  // (neg_a, A, neg_b, B, c_mod, C, reuse_a, reuse_b)
  return __builtin_amdgcn_wmma_f32_16x16x32_f16(false, a, false, b, (short)0, c,
                                                false, false);
}

// A fragment (16x32, f16) from row-major f16 buffer with leading dim ld.
// lane L: M = L&15 ; K base = (L>>4)*8 ; elems j<8 -> K=base+j, j>=8 -> K=base+16+(j-8)
__device__ __forceinline__ v16h a_frag_f16(const _Float16* base, int row0, int ld, int lane) {
  const _Float16* p = base + (size_t)(row0 + (lane & 15)) * ld + ((lane >> 4) << 3);
  v16h a;
#pragma unroll
  for (int j = 0; j < 8; ++j) { a[j] = p[j]; a[j + 8] = p[16 + j]; }
  return a;
}

// B fragment (32x16, f16) from row-major f16 buffer [K][N] with leading dim ld.
// lane L: N = L&15 ; K base = (L>>4)*8
__device__ __forceinline__ v16h b_frag_f16(const _Float16* base, int col0, int ld, int lane) {
  const _Float16* p = base + (size_t)((lane >> 4) << 3) * ld + col0 + (lane & 15);
  v16h b;
#pragma unroll
  for (int j = 0; j < 8; ++j) { b[j] = p[(size_t)j * ld]; b[j + 8] = p[(size_t)(16 + j) * ld]; }
  return b;
}

// ---------------- im2col (patch gather) ----------------
__global__ __launch_bounds__(256) void im2col_kernel(const float* __restrict__ x,
                                                     _Float16* __restrict__ patches) {
  int i = blockIdx.x * 256 + threadIdx.x;           // over MROWS*KPATCH
  if (i >= MROWS * KPATCH) return;
  int kk = i % KPATCH, m = i / KPATCH;
  int b = m >> 10, t = m & 1023;
  int py = t >> 5, px = t & 31;
  int c = kk >> 6, rem = kk & 63, ph = rem >> 3, pw = rem & 7;
  int hh = py * 8 + ph, ww = px * 8 + pw;
  patches[i] = (_Float16)x[(((size_t)b * 3 + c) * 256 + hh) * 256 + ww];
}

// w_conv [D,C,P,P] -> wt [K=192][N=768] fp32
__global__ __launch_bounds__(256) void wreshape_kernel(const float* __restrict__ wconv,
                                                       float* __restrict__ wt) {
  int i = blockIdx.x * 256 + threadIdx.x;           // over KPATCH*DMODEL
  if (i >= KPATCH * DMODEL) return;
  int d = i % DMODEL, kk = i / DMODEL;
  wt[i] = wconv[(size_t)d * KPATCH + kk];
}

// ---------------- LayerNorm: fp32 in -> f16 out ----------------
__global__ __launch_bounds__(256) void layernorm_kernel(const float* __restrict__ h,
                                                        const float* __restrict__ w,
                                                        const float* __restrict__ bb,
                                                        _Float16* __restrict__ y) {
  __shared__ float red[256];
  const int row = blockIdx.x, tid = threadIdx.x;
  const float* xr = h + (size_t)row * DMODEL;
  float v0 = xr[tid], v1 = xr[tid + 256], v2 = xr[tid + 512];
  red[tid] = v0 + v1 + v2;
  __syncthreads();
#pragma unroll
  for (int off = 128; off > 0; off >>= 1) {
    if (tid < off) red[tid] += red[tid + off];
    __syncthreads();
  }
  float mean = red[0] * (1.0f / DMODEL);
  __syncthreads();
  float d0 = v0 - mean, d1 = v1 - mean, d2 = v2 - mean;
  red[tid] = d0 * d0 + d1 * d1 + d2 * d2;
  __syncthreads();
#pragma unroll
  for (int off = 128; off > 0; off >>= 1) {
    if (tid < off) red[tid] += red[tid + off];
    __syncthreads();
  }
  float rstd = rsqrtf(red[0] * (1.0f / DMODEL) + 1e-5f);
  _Float16* yr = y + (size_t)row * DMODEL;
  yr[tid]       = (_Float16)(d0 * rstd * w[tid]       + bb[tid]);
  yr[tid + 256] = (_Float16)(d1 * rstd * w[tid + 256] + bb[tid + 256]);
  yr[tid + 512] = (_Float16)(d2 * rstd * w[tid + 512] + bb[tid + 512]);
}

// ---------------- tiled WMMA GEMM (double-buffered, async A staging) ----------------
// C[M,N] = A[M,K](f16) @ W[K,N](fp32 -> f16 in LDS) ; epilogue per MODE.
// Block tile 128x128, Kc=32, 256 threads = 8 waves, wave tile 32x64 (2x4 wmma frags).
// A tile: global->LDS via GLOBAL_LOAD_ASYNC_TO_LDS_B128 into the *next* buffer while the
// current buffer feeds WMMA. W tile: global->VGPR (overlaps compute) -> f16 ds_store into
// the next buffer (conversion prevents raw async copy).
// MODE 0: out16 = acc (+bias)            (qkv projection; bias==nullptr)
// MODE 1: out16 = gelu(acc + bias)       (MLP fc1, exact GELU)
// MODE 2: out32 += acc + bias            (residual adds: attn proj, MLP fc2)
// MODE 3: out32  = acc + bias + pos[tok] (patch embed)
static constexpr int AS_LD = 40;    // 32 + 8 pad (f16)
static constexpr int WS_LD = 136;   // 128 + 8 pad (f16)

template <int MODE>
__global__ __launch_bounds__(256) void gemm_kernel(
    const _Float16* __restrict__ A, const float* __restrict__ W,
    const float* __restrict__ bias, const float* __restrict__ pos,
    _Float16* __restrict__ out16, float* __restrict__ out32,
    int M, int N, int K) {
  __shared__ _Float16 As[2][128 * AS_LD];
  __shared__ _Float16 Ws[2][32 * WS_LD];

  const int tid = threadIdx.x;
  const int lane = tid & 31, wave = tid >> 5;
  const int wr = wave & 3;        // row group: 4 x 32 rows
  const int wc = wave >> 2;       // col group: 2 x 64 cols
  const int m0 = blockIdx.y * 128;
  const int n0 = blockIdx.x * 128;

  v8f acc[2][4];
  const v8f vz = {0.f, 0.f, 0.f, 0.f, 0.f, 0.f, 0.f, 0.f};
#pragma unroll
  for (int i = 0; i < 2; ++i)
#pragma unroll
    for (int j = 0; j < 4; ++j) acc[i][j] = vz;

  float4 wreg[4];

  // A tile 128x32 f16: 512 16B chunks, 2 per thread, async global->LDS
  auto stage_A_async = [&](int buf, int k0) {
    const _Float16* base = A + (size_t)m0 * K + k0;
#pragma unroll
    for (int u = 0; u < 2; ++u) {
      int i = tid + u * 256;
      int r = i >> 2, c = (i & 3) * 8;
      async_ld_b128(lds_off(&As[buf][r * AS_LD + c]), base,
                    (uint32_t)(((unsigned)r * (unsigned)K + (unsigned)c) * 2u));
    }
  };
  // W tile 32x128 fp32: 4 float4 per thread into registers
  auto load_W_regs = [&](int k0) {
#pragma unroll
    for (int u = 0; u < 4; ++u) {
      int i = tid + u * 256;
      int r = i >> 5, c = (i & 31) * 4;
      wreg[u] = *(const float4*)(W + (size_t)(k0 + r) * N + n0 + c);
    }
  };
  auto store_W_lds = [&](int buf) {
#pragma unroll
    for (int u = 0; u < 4; ++u) {
      int i = tid + u * 256;
      int r = i >> 5, c = (i & 31) * 4;
      _Float16* d = &Ws[buf][r * WS_LD + c];
      d[0] = (_Float16)wreg[u].x; d[1] = (_Float16)wreg[u].y;
      d[2] = (_Float16)wreg[u].z; d[3] = (_Float16)wreg[u].w;
    }
  };

  // prologue: tile 0 into buffer 0
  stage_A_async(0, 0);
  load_W_regs(0);
  store_W_lds(0);
  wait_async0();
  __syncthreads();

  const int nk = K >> 5;
  for (int t = 0; t < nk; ++t) {
    const int cur = t & 1, nxt = cur ^ 1;
    const bool more = (t + 1) < nk;
    if (more) {
      stage_A_async(nxt, (t + 1) * 32);   // async writes overlap WMMA below
      load_W_regs((t + 1) * 32);          // global loads overlap WMMA below
    }

    v16h af0 = a_frag_f16(As[cur], wr * 32,      AS_LD, lane);
    v16h af1 = a_frag_f16(As[cur], wr * 32 + 16, AS_LD, lane);
#pragma unroll
    for (int ct = 0; ct < 4; ++ct) {
      v16h bf = b_frag_f16(Ws[cur], wc * 64 + ct * 16, WS_LD, lane);
      acc[0][ct] = wmma16x16x32(af0, bf, acc[0][ct]);
      acc[1][ct] = wmma16x16x32(af1, bf, acc[1][ct]);
    }

    if (more) store_W_lds(nxt);           // different buffer: safe before barrier
    wait_async0();                        // this wave's async copies landed
    __syncthreads();                      // everyone's copies landed
  }

  // epilogue (C layout: row = r + 8*(lane>>4), col = lane&15)
  const int lr = lane & 15, lh = lane >> 4;
#pragma unroll
  for (int rt = 0; rt < 2; ++rt) {
#pragma unroll
    for (int ct = 0; ct < 4; ++ct) {
      const int gm0 = m0 + wr * 32 + rt * 16 + 8 * lh;
      const int gn = n0 + wc * 64 + ct * 16 + lr;
      const float bb = bias ? bias[gn] : 0.0f;
#pragma unroll
      for (int r = 0; r < 8; ++r) {
        const int gm = gm0 + r;
        float v = acc[rt][ct][r] + bb;
        if (MODE == 0) {
          out16[(size_t)gm * N + gn] = (_Float16)v;
        } else if (MODE == 1) {
          out16[(size_t)gm * N + gn] =
              (_Float16)(0.5f * v * (1.0f + erff(v * 0.70710678118f)));
        } else if (MODE == 2) {
          out32[(size_t)gm * N + gn] += v;
        } else {
          out32[(size_t)gm * N + gn] = v + pos[(size_t)(gm & 1023) * N + gn];
        }
      }
    }
  }
}

// ---------------- attention ----------------
// One workgroup = (batch b, head h, 64-query-row block). 256 threads = 8 waves.
// qkv: f16 [MROWS, 3*768]; q cols [h*64..), k cols [768+h*64..), v cols [1536+h*64..)
// Phase 1: S = (Q Kt) * scale into LDS (64 x 1024, stride SLD)
// Phase 2: exact softmax over rows (max, exp, sum, normalize) in LDS
// Phase 3: O = P V  -> o buffer f16 [MROWS, 768]
__global__ void attn_kernel(const _Float16* __restrict__ qkv, _Float16* __restrict__ o) {
  extern __shared__ float smem[];
  float* S    = smem;
  float* red  = smem + 64 * SLD;
  float* red2 = red + 256;

  const int tid = threadIdx.x, lane = tid & 31, wave = tid >> 5;
  int idx = blockIdx.x;
  const int rblk = idx & 15; idx >>= 4;          // 16 row blocks
  const int head = idx % HEADS;
  const int b = idx / HEADS;
  const int qoff = head * DH;
  const int koff = DMODEL + head * DH;
  const int voff = 2 * DMODEL + head * DH;
  const size_t base = (size_t)b * NTOK * (3 * DMODEL);
  const int row0 = rblk * 64;
  const float scale = 0.125f;                    // DH^-0.5

  const int wr = wave & 3;                       // 4 row tiles of 16
  const int wch = wave >> 2;                     // 2 key halves of 512

  // ---- phase 1: S = Q Kt * scale ----
  v16h qf[2];
  {
    const int m = row0 + wr * 16 + (lane & 15);
    const _Float16* p = qkv + base + (size_t)m * (3 * DMODEL) + qoff + ((lane >> 4) << 3);
#pragma unroll
    for (int f = 0; f < 2; ++f)
#pragma unroll
      for (int j = 0; j < 8; ++j) {
        qf[f][j]     = p[f * 32 + j];
        qf[f][j + 8] = p[f * 32 + 16 + j];
      }
  }
  for (int ct = 0; ct < 32; ++ct) {
    const int col0 = wch * 512 + ct * 16;
    v8f s = {0.f, 0.f, 0.f, 0.f, 0.f, 0.f, 0.f, 0.f};
#pragma unroll
    for (int kk = 0; kk < 2; ++kk) {
      // B frag: column n = key, K dim = dh (kk*32 .. +31)
      const int key = col0 + (lane & 15);
      const _Float16* p =
          qkv + base + (size_t)key * (3 * DMODEL) + koff + kk * 32 + ((lane >> 4) << 3);
      v16h bf;
#pragma unroll
      for (int j = 0; j < 8; ++j) { bf[j] = p[j]; bf[j + 8] = p[16 + j]; }
      s = wmma16x16x32(qf[kk], bf, s);
    }
    const int lr = lane & 15, lh = lane >> 4;
#pragma unroll
    for (int r = 0; r < 8; ++r)
      S[(wr * 16 + r + 8 * lh) * SLD + col0 + lr] = s[r] * scale;
  }
  __syncthreads();

  // ---- phase 2: softmax (each row handled by 4 threads x 256 cols) ----
  {
    const int row = tid >> 2, qq = tid & 3;
    float* Sr = S + row * SLD + qq * 256;
    float mx = -3.4e38f;
    for (int c = 0; c < 256; ++c) mx = fmaxf(mx, Sr[c]);
    red[row * 4 + qq] = mx;
    __syncthreads();
    mx = fmaxf(fmaxf(red[row * 4], red[row * 4 + 1]),
               fmaxf(red[row * 4 + 2], red[row * 4 + 3]));
    float sum = 0.f;
    for (int c = 0; c < 256; ++c) {
      float e = __expf(Sr[c] - mx);
      Sr[c] = e;
      sum += e;
    }
    red2[row * 4 + qq] = sum;
    __syncthreads();
    sum = red2[row * 4] + red2[row * 4 + 1] + red2[row * 4 + 2] + red2[row * 4 + 3];
    const float inv = 1.0f / sum;
    for (int c = 0; c < 256; ++c) Sr[c] *= inv;
  }
  __syncthreads();

  // ---- phase 3: O = P V ----
  const int cho = wave >> 2;                     // 2 dh halves of 32
  v8f oacc[2];
  oacc[0] = (v8f){0.f, 0.f, 0.f, 0.f, 0.f, 0.f, 0.f, 0.f};
  oacc[1] = oacc[0];
  for (int kk = 0; kk < NTOK; kk += 32) {
    // A frag = softmaxed P from LDS (f32 -> f16 on load)
    v16h pf;
    {
      const float* p = S + (size_t)(wr * 16 + (lane & 15)) * SLD + kk + ((lane >> 4) << 3);
#pragma unroll
      for (int j = 0; j < 8; ++j) { pf[j] = (_Float16)p[j]; pf[j + 8] = (_Float16)p[16 + j]; }
    }
#pragma unroll
    for (int t2 = 0; t2 < 2; ++t2) {
      const int n = cho * 32 + t2 * 16 + (lane & 15);
      const int kb = kk + ((lane >> 4) << 3);
      const _Float16* p = qkv + base + voff + n;
      v16h vf;
#pragma unroll
      for (int j = 0; j < 8; ++j) {
        vf[j]     = p[(size_t)(kb + j) * (3 * DMODEL)];
        vf[j + 8] = p[(size_t)(kb + 16 + j) * (3 * DMODEL)];
      }
      oacc[t2] = wmma16x16x32(pf, vf, oacc[t2]);
    }
  }
  const int lr = lane & 15, lh = lane >> 4;
#pragma unroll
  for (int t2 = 0; t2 < 2; ++t2)
#pragma unroll
    for (int r = 0; r < 8; ++r) {
      const int gr = b * NTOK + row0 + wr * 16 + r + 8 * lh;
      const int gc = head * DH + cho * 32 + t2 * 16 + lr;
      o[(size_t)gr * DMODEL + gc] = (_Float16)oacc[t2][r];
    }
}

// ---------------- host orchestration ----------------
extern "C" void kernel_launch(void* const* d_in, const int* in_sizes, int n_in,
                              void* d_out, int out_size, void* d_ws, size_t ws_size,
                              hipStream_t stream) {
  (void)in_sizes; (void)n_in; (void)out_size; (void)ws_size;
  const float* x       = (const float*)d_in[0];
  const float* w_conv  = (const float*)d_in[1];
  const float* b_conv  = (const float*)d_in[2];
  const float* pos_emb = (const float*)d_in[3];
  const float* ln1_w   = (const float*)d_in[4];
  const float* ln1_b   = (const float*)d_in[5];
  const float* qkv_w   = (const float*)d_in[6];
  const float* out_w   = (const float*)d_in[7];
  const float* out_b   = (const float*)d_in[8];
  const float* ln2_w   = (const float*)d_in[9];
  const float* ln2_b   = (const float*)d_in[10];
  const float* w1      = (const float*)d_in[11];
  const float* b1      = (const float*)d_in[12];
  const float* w2      = (const float*)d_in[13];
  const float* b2      = (const float*)d_in[14];

  float* h = (float*)d_out;                      // residual stream [8192, 768] fp32

  // workspace carve (deterministic layout)
  char* ws = (char*)d_ws;
  size_t off = 0;
  auto carve = [&](size_t bytes) {
    void* p = ws + off;
    off = (off + bytes + 255) & ~(size_t)255;
    return p;
  };
  _Float16* patches = (_Float16*)carve((size_t)MROWS * KPATCH * 2);
  float*    wt      = (float*)carve((size_t)KPATCH * DMODEL * 4);
  _Float16* ybuf    = (_Float16*)carve((size_t)MROWS * DMODEL * 2);
  _Float16* qkvbuf  = (_Float16*)carve((size_t)MROWS * 3 * DMODEL * 2);
  _Float16* obuf    = (_Float16*)carve((size_t)MROWS * DMODEL * 2);
  _Float16* mlpbuf  = (_Float16*)carve((size_t)MROWS * MLPDIM * 2);

  const dim3 blk(256);

  // patch embed: im2col + reshape, then GEMM with pos-emb epilogue
  im2col_kernel<<<(MROWS * KPATCH + 255) / 256, blk, 0, stream>>>(x, patches);
  wreshape_kernel<<<(KPATCH * DMODEL + 255) / 256, blk, 0, stream>>>(w_conv, wt);
  gemm_kernel<3><<<dim3(DMODEL / 128, MROWS / 128), blk, 0, stream>>>(
      patches, wt, b_conv, pos_emb, nullptr, h, MROWS, DMODEL, KPATCH);

  // opt into >64KB dynamic LDS for the attention kernel (320KB/WGP on CDNA5)
  (void)hipFuncSetAttribute((const void*)attn_kernel,
                            hipFuncAttributeMaxDynamicSharedMemorySize, ATTN_LDS_BYTES);

  for (int l = 0; l < DEPTH; ++l) {
    const float* l1w = ln1_w + (size_t)l * DMODEL;
    const float* l1b = ln1_b + (size_t)l * DMODEL;
    const float* qw  = qkv_w + (size_t)l * DMODEL * 3 * DMODEL;
    const float* ow  = out_w + (size_t)l * DMODEL * DMODEL;
    const float* ob  = out_b + (size_t)l * DMODEL;
    const float* l2w = ln2_w + (size_t)l * DMODEL;
    const float* l2b = ln2_b + (size_t)l * DMODEL;
    const float* W1  = w1 + (size_t)l * DMODEL * MLPDIM;
    const float* B1  = b1 + (size_t)l * MLPDIM;
    const float* W2  = w2 + (size_t)l * MLPDIM * DMODEL;
    const float* B2  = b2 + (size_t)l * DMODEL;

    // y = LN(h) -> f16
    layernorm_kernel<<<MROWS, blk, 0, stream>>>(h, l1w, l1b, ybuf);
    // qkv = y @ qkv_w (no bias)
    gemm_kernel<0><<<dim3(3 * DMODEL / 128, MROWS / 128), blk, 0, stream>>>(
        ybuf, qw, nullptr, nullptr, qkvbuf, nullptr, MROWS, 3 * DMODEL, DMODEL);
    // attention
    attn_kernel<<<BATCH * HEADS * (NTOK / 64), blk, ATTN_LDS_BYTES, stream>>>(qkvbuf, obuf);
    // h += o @ out_w + out_b
    gemm_kernel<2><<<dim3(DMODEL / 128, MROWS / 128), blk, 0, stream>>>(
        obuf, ow, ob, nullptr, nullptr, h, MROWS, DMODEL, DMODEL);
    // y = LN(h) -> f16
    layernorm_kernel<<<MROWS, blk, 0, stream>>>(h, l2w, l2b, ybuf);
    // mlp = gelu(y @ w1 + b1)
    gemm_kernel<1><<<dim3(MLPDIM / 128, MROWS / 128), blk, 0, stream>>>(
        ybuf, W1, B1, nullptr, mlpbuf, nullptr, MROWS, MLPDIM, DMODEL);
    // h += mlp @ w2 + b2
    gemm_kernel<2><<<dim3(DMODEL / 128, MROWS / 128), blk, 0, stream>>>(
        mlpbuf, W2, B2, nullptr, nullptr, h, MROWS, DMODEL, MLPDIM);
  }
}
